// LoRACacheModule_68513318305908
// MI455X (gfx1250) — matrix operand order repaired and checked by reading, not compiled, not run on hardware
//
#include <hip/hip_runtime.h>

typedef _Float16 f16;
typedef __attribute__((ext_vector_type(16))) _Float16 v16h;
typedef __attribute__((ext_vector_type(8)))  float    v8f;
typedef __attribute__((ext_vector_type(4)))  float    v4f;
typedef __attribute__((ext_vector_type(2)))  float    v2f;

#define BB 16
#define HH 192
#define WW 192
#define CC 128
#define RR 16
#define RKK 8
#define NCACHE 5
#define HWSZ (HH*WW)            /* 36864 */
#define TOK_TOTAL (BB*HWSZ)     /* 589824 */
#define SCALEV (1.0f/8.0f)
#define THRESHV 0.7f
#define EPSV 1e-8f

/* fragment table: B-layout f16 fragments, 512 halves (1KB) each */
#define F_A1 0    /* 4 frags : x@A1pad  (K tiles of 32, N padded 8->16)  */
#define F_B1 4    /* 1 frag  : B1*SCALE (K 8->32 pad, N=16)              */
#define F_A2 5    /* 1 frag  : A2 (K 16->32 pad, N 8->16 pad)            */
#define F_B2 6    /* 8 frags : B2*SCALE (K 8->32 pad, N tiles of 16)     */
#define F_W1 14   /* 40 frags: W1 [144->160 pad][128], kt 0..4, nt 0..7  */
#define F_W2 54   /* 32 frags: W2 [128][128], kt 0..3, nt 0..7           */
#define NFRAG 86

union U16x { v16h v; v4f q[2]; };
union U4x  { f16 h[4]; v2f f; };

static __device__ __forceinline__ v8f vzero8() {
  v8f z = {0.f,0.f,0.f,0.f,0.f,0.f,0.f,0.f};
  return z;
}

static __device__ __forceinline__ v8f wmma16(v16h a, v16h b, v8f c) {
  /* D = A(16x32 f16) * B(32x16 f16) + C(16x16 f32) */
  return __builtin_amdgcn_wmma_f32_16x16x32_f16(false, a, false, b, (short)0, c,
                                                false, false);
}

/* load a pre-swizzled weight fragment: lane's 16 halves are contiguous (32B) */
static __device__ __forceinline__ v16h load_wfrag(const f16* wf, int f, int lane) {
  const v4f* p = (const v4f*)(wf + (f << 9) + (lane << 4));
  U16x u; u.q[0] = p[0]; u.q[1] = p[1];
  return u.v;
}

/* build A fragment (16x32 f16) from a row-major LDS tile, k-tile kt.
   lane<16: row=lane,    K = {0..7, 16..23} of the tile
   lane>=16: row=lane-16, K = {8..15, 24..31}                     */
static __device__ __forceinline__ v16h load_afrag(const f16* t, int ldk, int kt, int lane) {
  int r  = lane & 15;
  int ko = (lane >> 4) * 8;
  const f16* p = t + r * ldk + kt * 32 + ko;
  U16x u;
  u.q[0] = *(const v4f*)(p);        /* halves K=ko..ko+7    */
  u.q[1] = *(const v4f*)(p + 16);   /* halves K=ko+16..+23  */
  return u.v;
}

/* store a 16x16 f32 accumulator (C layout) to a row-major f16 LDS tile */
static __device__ __forceinline__ void stage_c(f16* s, int ldk, v8f c, int lane) {
  int n  = lane & 15;
  int mo = (lane >> 4) * 8;
#pragma unroll
  for (int v = 0; v < 8; ++v) s[(mo + v) * ldk + n] = (f16)c[v];
}

/* load 16 token rows (16x128 f32) -> f16 LDS tile */
static __device__ __forceinline__ void load_x_tile(const float* gx, f16* lx, int lane) {
  int r  = lane & 15;
  int hf = lane >> 4;
  const float* src = gx + r * CC + hf * 64;
  f16*         dst = lx + r * CC + hf * 64;
#pragma unroll
  for (int j = 0; j < 16; ++j) {
    v4f t = *(const v4f*)(src + j * 4);
    U4x u;
    u.h[0] = (f16)t.x; u.h[1] = (f16)t.y; u.h[2] = (f16)t.z; u.h[3] = (f16)t.w;
    *(v2f*)(dst + j * 4) = u.f;   /* ds_store_b64 */
  }
}

/* full LoRA chain via WMMA: xl[8] = x_lora (C layout per n-tile), down = lora_down */
static __device__ __forceinline__ void lora_forward(const f16* wf, const f16* lx,
                                                    f16* ls, int lane,
                                                    v8f xl[8], v8f& down) {
  v8f acc = vzero8();
#pragma unroll
  for (int kt = 0; kt < 4; ++kt)
    acc = wmma16(load_afrag(lx, CC, kt, lane), load_wfrag(wf, F_A1 + kt, lane), acc);

  stage_c(ls, 32, acc, lane);
  asm volatile("s_wait_dscnt 0" ::: "memory");
  down = wmma16(load_afrag(ls, 32, 0, lane), load_wfrag(wf, F_B1, lane), vzero8());

  stage_c(ls, 32, down, lane);
  asm volatile("s_wait_dscnt 0" ::: "memory");
  v8f t2 = wmma16(load_afrag(ls, 32, 0, lane), load_wfrag(wf, F_A2, lane), vzero8());

  stage_c(ls, 32, t2, lane);
  asm volatile("s_wait_dscnt 0" ::: "memory");
  v16h a3 = load_afrag(ls, 32, 0, lane);
#pragma unroll
  for (int nt = 0; nt < 8; ++nt)
    xl[nt] = wmma16(a3, load_wfrag(wf, F_B2 + nt, lane), vzero8());
}

/* ---- phase 0: pre-swizzle weights into B-matrix fragment layout ---- */
__global__ void pack_kernel(const float* A1, const float* B1, const float* A2,
                            const float* B2, const float* W1, const float* W2,
                            f16* wf) {
  int i = blockIdx.x * blockDim.x + threadIdx.x;
  if (i >= NFRAG * 512) return;
  int f = i >> 9, lidx = i & 511;
  int lane = lidx >> 4, h = lidx & 15;
  int nn = lane & 15;
  int k  = ((h >> 3) << 4) + ((lane >> 4) << 3) + (h & 7);  /* B layout: lanes 0-15 K={0..7,16..23}, 16-31 K={8..15,24..31} */
  float v = 0.f;
  if (f < 4)       { int K = f * 32 + k; if (nn < RKK) v = A1[K * RKK + nn]; }
  else if (f == 4) { if (k < RKK) v = B1[k * RR + nn] * SCALEV; }
  else if (f == 5) { if (k < RR && nn < RKK) v = A2[k * RKK + nn]; }
  else if (f < 14) { int nt = f - 6; if (k < RKK) v = B2[k * CC + nt * 16 + nn] * SCALEV; }
  else if (f < 54) { int g = f - F_W1, kt = g >> 3, nt = g & 7;
                     int K = kt * 32 + k; if (K < CC + RR) v = W1[K * CC + nt * 16 + nn]; }
  else             { int g = f - F_W2, kt = g >> 3, nt = g & 7;
                     v = W2[(kt * 32 + k) * CC + nt * 16 + nn]; }
  wf[(f << 9) + (lane << 4) + h] = (f16)v;
}

/* ---- phase 0b: cache squared norms per (n,c) ---- */
__global__ void cn2_kernel(const float* cache, float* cn2) {
  int c = threadIdx.x, n = blockIdx.x, chunk = blockIdx.y;
  float acc = 0.f;
  for (int r = chunk * 576; r < (chunk + 1) * 576; ++r) {
    float v = cache[((size_t)n * HWSZ + r) * CC + c];
    acc += v * v;
  }
  atomicAdd(&cn2[n * CC + c], acc);
}

/* ---- phase 1: x_lora (WMMA) + num/xn2 reductions ---- */
__global__ __launch_bounds__(128, 1)
void sim_reduce_kernel(const float* x, const float* cache, const f16* wf,
                       float* num, float* xn2) {
  __shared__ f16 s_x[4][16 * CC];
  __shared__ f16 s_s[4][16 * 32];
  int w = threadIdx.x >> 5, lane = threadIdx.x & 31;
  int b = blockIdx.y;
  int wid = blockIdx.x * 4 + w;            /* 0..575 waves per batch */
  int n_ = lane & 15, hf = lane >> 4;
  f16 *lx = s_x[w], *ls = s_s[w];
  v4f z4 = {0.f, 0.f, 0.f, 0.f};
  *(v4f*)(ls + n_ * 32 + 16 + hf * 8) = z4;   /* zero pad K=16..31 once */

  float pnum[NCACHE][8];
  float pxn[8];
#pragma unroll
  for (int ci = 0; ci < NCACHE; ++ci)
#pragma unroll
    for (int nt = 0; nt < 8; ++nt) pnum[ci][nt] = 0.f;
#pragma unroll
  for (int nt = 0; nt < 8; ++nt) pxn[nt] = 0.f;

  for (int t = 0; t < 4; ++t) {
    size_t tile = (size_t)wid * 4 + t;                /* 0..2303 */
    size_t T = (size_t)b * HWSZ + tile * 16;
    load_x_tile(x + T * CC, lx, lane);
    asm volatile("s_wait_dscnt 0" ::: "memory");
    v8f xl[8]; v8f down;
    lora_forward(wf, lx, ls, lane, xl, down);
    size_t hwb = tile * 16;
#pragma unroll
    for (int ci = 0; ci < NCACHE; ++ci) {
      const float* cp = cache + ((size_t)ci * HWSZ + hwb) * CC;
#pragma unroll
      for (int nt = 0; nt < 8; ++nt) {
        float acc = 0.f;
#pragma unroll
        for (int v = 0; v < 8; ++v) {
          float xv = xl[nt][v];
          float cv = cp[(size_t)(v + hf * 8) * CC + nt * 16 + n_];
          acc += xv * cv;
          if (ci == 0) pxn[nt] += xv * xv;
        }
        pnum[ci][nt] += acc;
      }
    }
  }
  /* combine lane halves (same channel, different rows) then atomics */
#pragma unroll
  for (int ci = 0; ci < NCACHE; ++ci)
#pragma unroll
    for (int nt = 0; nt < 8; ++nt) {
      float s = pnum[ci][nt] + __shfl_xor(pnum[ci][nt], 16, 32);
      if (hf == 0) atomicAdd(&num[((b * NCACHE) + ci) * CC + nt * 16 + n_], s);
    }
#pragma unroll
  for (int nt = 0; nt < 8; ++nt) {
    float s = pxn[nt] + __shfl_xor(pxn[nt], 16, 32);
    if (hf == 0) atomicAdd(&xn2[b * CC + nt * 16 + n_], s);
  }
}

/* ---- phase 2: per-batch argmax over cosine sims ---- */
__global__ void argmax_kernel(const float* num, const float* xn2, const float* cn2,
                              int* bidx, int* bmask) {
  __shared__ float ssims[BB * NCACHE];
  int t = threadIdx.x;
  if (t < BB * NCACHE) {
    int b = t / NCACHE, nn = t % NCACHE;
    float s = 0.f;
    for (int c = 0; c < CC; ++c) {
      float xn = sqrtf(xn2[b * CC + c]);
      float cn = sqrtf(cn2[nn * CC + c]);
      float d = fmaxf(xn, EPSV) * fmaxf(cn, EPSV);
      s += num[(b * NCACHE + nn) * CC + c] / d;
    }
    ssims[t] = s / (float)CC;
  }
  __syncthreads();
  if (t < BB) {
    float best = -1e30f; int bi = 0;
    for (int nn = 0; nn < NCACHE; ++nn) {
      float s = ssims[t * NCACHE + nn];
      if (s > best) { best = s; bi = nn; }     /* strict > keeps first max */
    }
    bidx[t] = bi;
    bmask[t] = (best > THRESHV) ? 1 : 0;
  }
}

/* ---- phase 3: recompute lora, blend, fusion MLP via WMMA, select ---- */
__global__ __launch_bounds__(128, 1)
void fuse_kernel(const float* x, const float* cache, const float* b1v,
                 const float* b2v, const float* cwp, const f16* wf,
                 const int* bidx, const int* bmask, float* out) {
  __shared__ f16 s_x[4][16 * CC];      /* 4KB/wave  */
  __shared__ f16 s_s[4][16 * 32];      /* 1KB/wave  */
  __shared__ f16 s_cat[4][16 * 160];   /* 5KB/wave  */
  __shared__ f16 s_h[4][16 * CC];      /* 4KB/wave  */
  int w = threadIdx.x >> 5, lane = threadIdx.x & 31;
  size_t T = ((size_t)blockIdx.x * 4 + w) * 16;
  int b = (int)(T / HWSZ);             /* uniform per wave (HWSZ % 16 == 0) */
  int n_ = lane & 15, hf = lane >> 4;
  f16 *lx = s_x[w], *ls = s_s[w];
  v4f z4 = {0.f, 0.f, 0.f, 0.f};
  *(v4f*)(ls + n_ * 32 + 16 + hf * 8) = z4;

  load_x_tile(x + T * CC, lx, lane);
  asm volatile("s_wait_dscnt 0" ::: "memory");
  v8f xl[8]; v8f down;
  lora_forward(wf, lx, ls, lane, xl, down);

  if (bmask[b]) {                      /* uniform branch: EXEC stays all-1s */
    float cw = 1.f / (1.f + __expf(-cwp[0]));
    int ci = bidx[b];
    size_t hwb = T - (size_t)b * HWSZ;
    const float* bp = cache + ((size_t)ci * HWSZ + hwb) * CC;
    f16* lc = s_cat[w];
    *(v4f*)(lc + n_ * 160 + 144 + hf * 8) = z4;   /* zero K=144..159 */
    /* blend -> cat[:,0:128] */
#pragma unroll
    for (int nt = 0; nt < 8; ++nt)
#pragma unroll
      for (int v = 0; v < 8; ++v) {
        int m = v + hf * 8, c = nt * 16 + n_;
        float bv = bp[(size_t)m * CC + c];
        float bl = cw * xl[nt][v] + (1.f - cw) * bv;
        lc[m * 160 + c] = (f16)bl;
      }
    /* down -> cat[:,128:144] */
#pragma unroll
    for (int v = 0; v < 8; ++v)
      lc[(v + hf * 8) * 160 + 128 + n_] = (f16)down[v];
    asm volatile("s_wait_dscnt 0" ::: "memory");

    v16h acat[5];
#pragma unroll
    for (int kt = 0; kt < 5; ++kt) acat[kt] = load_afrag(lc, 160, kt, lane);
    f16* lh = s_h[w];
#pragma unroll
    for (int nt = 0; nt < 8; ++nt) {
      v8f hc = vzero8();
#pragma unroll
      for (int kt = 0; kt < 5; ++kt)
        hc = wmma16(acat[kt], load_wfrag(wf, F_W1 + kt * 8 + nt, lane), hc);
      float bias = b1v[nt * 16 + n_];
#pragma unroll
      for (int v = 0; v < 8; ++v) {
        float hv = hc[v] + bias;
        hv = hv / (1.f + __expf(-hv));          /* silu */
        lh[(v + hf * 8) * CC + nt * 16 + n_] = (f16)hv;
      }
    }
    asm volatile("s_wait_dscnt 0" ::: "memory");

    v16h ah[4];
#pragma unroll
    for (int kt = 0; kt < 4; ++kt) ah[kt] = load_afrag(lh, CC, kt, lane);
#pragma unroll
    for (int nt = 0; nt < 8; ++nt) {
      v8f oc = vzero8();
#pragma unroll
      for (int kt = 0; kt < 4; ++kt)
        oc = wmma16(ah[kt], load_wfrag(wf, F_W2 + kt * 8 + nt, lane), oc);
      float bias = b2v[nt * 16 + n_];
#pragma unroll
      for (int v = 0; v < 8; ++v)
        out[(T + v + hf * 8) * CC + nt * 16 + n_] = oc[v] + bias;
    }
  } else {
#pragma unroll
    for (int nt = 0; nt < 8; ++nt)
#pragma unroll
      for (int v = 0; v < 8; ++v)
        out[(T + v + hf * 8) * CC + nt * 16 + n_] = xl[nt][v];
  }
}

extern "C" void kernel_launch(void* const* d_in, const int* in_sizes, int n_in,
                              void* d_out, int out_size, void* d_ws, size_t ws_size,
                              hipStream_t stream) {
  const float* x     = (const float*)d_in[0];
  const float* A1    = (const float*)d_in[1];
  const float* B1    = (const float*)d_in[2];
  const float* A2    = (const float*)d_in[3];
  const float* B2    = (const float*)d_in[4];
  const float* W1    = (const float*)d_in[5];
  const float* b1v   = (const float*)d_in[6];
  const float* W2    = (const float*)d_in[7];
  const float* b2v   = (const float*)d_in[8];
  const float* cw    = (const float*)d_in[9];
  const float* cache = (const float*)d_in[10];

  char* ws = (char*)d_ws;
  f16*   wf   = (f16*)ws;                       /* 86*1024 = 88064 B    */
  float* num  = (float*)(ws + 90112);           /* 16*5*128*4 = 40960 B */
  float* xn2  = (float*)(ws + 131072);          /* 16*128*4   = 8192 B  */
  float* cn2  = (float*)(ws + 139264);          /* 5*128*4    = 2560 B  */
  int*   bidx = (int*)(ws + 141824);            /* 64 B                 */
  int*   bmsk = (int*)(ws + 141888);            /* 64 B                 */

  /* zero the accumulators every launch (harness poisons ws) */
  hipMemsetAsync(ws + 90112, 0, 141952 - 90112, stream);

  pack_kernel<<<172, 256, 0, stream>>>(A1, B1, A2, B2, W1, W2, wf);
  cn2_kernel<<<dim3(NCACHE, 64), 128, 0, stream>>>(cache, cn2);
  sim_reduce_kernel<<<dim3(144, BB), 128, 0, stream>>>(x, cache, wf, num, xn2);
  argmax_kernel<<<1, 128, 0, stream>>>(num, xn2, cn2, bidx, bmsk);
  fuse_kernel<<<TOK_TOTAL / 64, 128, 0, stream>>>(x, cache, b1v, b2v, cw, wf,
                                                  bidx, bmsk, (float*)d_out);
}
